// PointNet_v4_6579889897775
// MI455X (gfx1250) — matrix operand order repaired and checked by reading, not compiled, not run on hardware
//
#include <hip/hip_runtime.h>

#define NPTS 16384
#define NB 16
#define PSTR 68              // per-point LDS stride in dwords (64 pairs + 4 pad: bank-rotating)
#define INV 0xFFFFFFFFu

typedef __attribute__((ext_vector_type(16))) _Float16 v16h;
typedef __attribute__((ext_vector_type(8)))  float    v8f;

union ABReg { v16h v; unsigned u[8]; uint4 q[2]; _Float16 f[16]; };

__device__ __forceinline__ unsigned pack2(float a, float b) {
    union { _Float16 f[2]; unsigned u; } t;
    t.f[0] = (_Float16)a; t.f[1] = (_Float16)b;
    return t.u;
}

__device__ __forceinline__ v16h loadA(const unsigned short* base, int tile, int lane) {
    ABReg r;
    const uint4* p = (const uint4*)(base + (size_t)tile * 512 + lane * 16);
    r.q[0] = p[0]; r.q[1] = p[1];
    return r.v;
}

// ---------------- phase-1 generic layer: LDS(bufIn) -> LDS(bufOut) ----------------
template<int KT, int MT>
__device__ __forceinline__ void layer_fw(const unsigned* bufIn, unsigned* bufOut,
                                         const unsigned short* wb, const float* bias,
                                         int q, int lane) {
    const bool hi = (lane >= 16);
    const int hb = hi ? 8 : 0;
    ABReg Bt[KT];
#pragma unroll
    for (int kt = 0; kt < KT; ++kt) {           // 2x ds_load_b128 per k-tile
        const uint4* src = (const uint4*)&bufIn[q * PSTR + kt * 16 + hb];
        Bt[kt].q[0] = src[0];
        Bt[kt].q[1] = src[1];
    }
#pragma unroll
    for (int mt = 0; mt < MT; ++mt) {
        int chB = mt * 16 + hb;
        v8f c;
#pragma unroll
        for (int i = 0; i < 8; ++i) c[i] = bias[chB + i];
#pragma unroll
        for (int kt = 0; kt < KT; ++kt) {
            v16h a = loadA(wb, mt * KT + kt, lane);
            c = __builtin_amdgcn_wmma_f32_16x16x32_f16(false, a, false, Bt[kt].v,
                                                       (short)0, c, false, false);
        }
        uint4 o4;
        o4.x = pack2(fmaxf(c[0], 0.f), fmaxf(c[1], 0.f));
        o4.y = pack2(fmaxf(c[2], 0.f), fmaxf(c[3], 0.f));
        o4.z = pack2(fmaxf(c[4], 0.f), fmaxf(c[5], 0.f));
        o4.w = pack2(fmaxf(c[6], 0.f), fmaxf(c[7], 0.f));
        *(uint4*)&bufOut[q * PSTR + mt * 8 + (hi ? 4 : 0)] = o4;   // 1x ds_store_b128
    }
}

// ---------------- fused local-MLP + max/argmax/group-max kernel ----------------
__global__ __launch_bounds__(256) void k_main(
    const float* __restrict__ pts, const unsigned* __restrict__ cnt,
    const unsigned* __restrict__ perm, const unsigned short* __restrict__ wpack,
    const float* __restrict__ bl0, const float* __restrict__ bl1,
    const float* __restrict__ bl2, const float* __restrict__ bl3,
    const float* __restrict__ bl4,
    unsigned long long* __restrict__ gacc, unsigned* __restrict__ grp)
{
    int bid = blockIdx.x;
    int tilei = bid & 127, g = (bid >> 7) & 3, b = bid >> 9;
    unsigned n_in = cnt[b * 4 + g];
    unsigned tileStart = (unsigned)tilei * 128u;
    if (tileStart >= n_in) return;

    __shared__ unsigned sb0[128 * PSTR];
    __shared__ unsigned sb1[128 * PSTR];
    __shared__ unsigned s_p[128];

    int t = threadIdx.x, w = t >> 5, lane = t & 31, ln = lane & 15;
    bool hi = (lane >= 16);
    const int hb = hi ? 8 : 0;

    if (t < 128) {
        unsigned slot = tileStart + (unsigned)t;
        s_p[t] = (slot < n_in) ? perm[(size_t)(b * 4 + g) * NPTS + slot] : INV;
    }

    // prefetch this wave's L4 weight slice (32KB) while L0-L3 run
    {
        const char* slice = (const char*)(wpack + 18432 + (size_t)(w * 8) * 4 * 512);
#pragma unroll
        for (int i = 0; i < 8; ++i)
            __builtin_prefetch(slice + (size_t)(i * 32 + lane) * 128, 0, 3);
    }
    __syncthreads();

    int q = w * 16 + ln;

    // ---- L0: 7 -> 64 (B built from gathered, group-sorted input points) ----
    ABReg B0;
#pragma unroll
    for (int j = 0; j < 8; ++j) B0.u[j] = 0u;
    if (!hi) {
        unsigned p = s_p[q];
        unsigned pp = (p == INV) ? 0u : p;
        const float* P = pts + (size_t)b * 7 * NPTS + pp;
#pragma unroll
        for (int c7 = 0; c7 < 7; ++c7) B0.f[c7] = (_Float16)P[(size_t)c7 * NPTS];
    }
#pragma unroll
    for (int mt = 0; mt < 4; ++mt) {
        int chB = mt * 16 + hb;
        v8f c;
#pragma unroll
        for (int i = 0; i < 8; ++i) c[i] = bl0[chB + i];
        v16h a = loadA(wpack + 0, mt, lane);
        c = __builtin_amdgcn_wmma_f32_16x16x32_f16(false, a, false, B0.v,
                                                   (short)0, c, false, false);
        uint4 o4;
        o4.x = pack2(fmaxf(c[0], 0.f), fmaxf(c[1], 0.f));
        o4.y = pack2(fmaxf(c[2], 0.f), fmaxf(c[3], 0.f));
        o4.z = pack2(fmaxf(c[4], 0.f), fmaxf(c[5], 0.f));
        o4.w = pack2(fmaxf(c[6], 0.f), fmaxf(c[7], 0.f));
        *(uint4*)&sb0[q * PSTR + mt * 8 + (hi ? 4 : 0)] = o4;
    }
    __syncthreads();
    layer_fw<2, 4>(sb0, sb1, wpack + 2048, bl1, q, lane);   // L1 64->64
    __syncthreads();
    layer_fw<2, 4>(sb1, sb0, wpack + 6144, bl2, q, lane);   // L2 64->64
    __syncthreads();
    layer_fw<2, 8>(sb0, sb1, wpack + 10240, bl3, q, lane);  // L3 64->128
    __syncthreads();

    // ---- L4: 128 -> 1024, wave owns 128 output channels, fused reduction ----
    const unsigned short* w4 = wpack + 18432;
    unsigned pv[8];
#pragma unroll
    for (int nt = 0; nt < 8; ++nt) pv[nt] = s_p[nt * 16 + ln];

    for (int mt = 0; mt < 8; ++mt) {
        int mtile = w * 8 + mt;
        v16h A4[4];
#pragma unroll
        for (int kt = 0; kt < 4; ++kt) A4[kt] = loadA(w4, mtile * 4 + kt, lane);
        int chB = mtile * 16 + hb;
        float bv[8];
#pragma unroll
        for (int i = 0; i < 8; ++i) bv[i] = bl4[chB + i];
        float rv[8]; unsigned ri[8];
#pragma unroll
        for (int i = 0; i < 8; ++i) { rv[i] = -1.f; ri[i] = INV; }

#pragma unroll
        for (int nt = 0; nt < 8; ++nt) {
            int qq = nt * 16 + ln;
            v8f c;
#pragma unroll
            for (int i = 0; i < 8; ++i) c[i] = bv[i];
#pragma unroll
            for (int kt = 0; kt < 4; ++kt) {
                ABReg Bt;
                const uint4* src = (const uint4*)&sb1[qq * PSTR + kt * 16 + hb];
                Bt.q[0] = src[0];
                Bt.q[1] = src[1];
                c = __builtin_amdgcn_wmma_f32_16x16x32_f16(false, A4[kt], false, Bt.v,
                                                           (short)0, c, false, false);
            }
            unsigned p = pv[nt];
            bool valid = (p != INV);
#pragma unroll
            for (int i = 0; i < 8; ++i) {
                float v = fmaxf(c[i], 0.f);                 // ReLU fused with reduction
                if (valid && (v > rv[i] || (v == rv[i] && p < ri[i]))) { rv[i] = v; ri[i] = p; }
            }
        }
        // 16-lane tree reduction (XOR masks < 16 keep the two channel-halves separate)
#pragma unroll
        for (int m = 1; m < 16; m <<= 1) {
#pragma unroll
            for (int i = 0; i < 8; ++i) {
                float ov = __shfl_xor(rv[i], m, 32);
                unsigned oi = (unsigned)__shfl_xor((int)ri[i], m, 32);
                if (ov > rv[i] || (ov == rv[i] && oi < ri[i])) { rv[i] = ov; ri[i] = oi; }
            }
        }
        if (ln == 0) {
#pragma unroll
            for (int i = 0; i < 8; ++i) {
                if (rv[i] >= 0.f) {
                    int ch = chB + i;
                    unsigned vb = __float_as_uint(rv[i]);   // >=0 -> monotonic as uint
                    unsigned long long key =
                        ((unsigned long long)vb << 32) | (unsigned long long)(unsigned)(~ri[i]);
                    atomicMax(&gacc[(size_t)b * 1024 + ch], key);
                    atomicMax(&grp[g * 1024 + ch], vb);
                }
            }
        }
    }
}

// ---------------- small helper kernels ----------------
__global__ void k_init(unsigned long long* gacc, unsigned* grp, unsigned* cnt) {
    int i = blockIdx.x * blockDim.x + threadIdx.x;
    if (i < 16384) gacc[i] = 0ull;
    if (i < 4096)  grp[i] = 0u;
    if (i < 64)    cnt[i] = 0u;
}

// pack fp32 weights into f16 WMMA A-tile lane order
__global__ void k_pack(const float* __restrict__ W, unsigned short* __restrict__ dst,
                       int Cout, int Cin, int KT) {
    int tid = blockIdx.x * blockDim.x + threadIdx.x;
    int total = (Cout >> 4) * KT * 512;
    if (tid >= total) return;
    int tile = tid >> 9, r = tid & 511, lane = r >> 4, h = r & 15;
    int mt = tile / KT, kt = tile % KT;
    int M = mt * 16 + (lane & 15);
    int K = kt * 32 + h + ((lane < 16) ? 0 : 16);
    float v = (K < Cin) ? W[(size_t)M * Cin + K] : 0.f;
    union { _Float16 f; unsigned short u; } cv; cv.f = (_Float16)v;
    dst[tid] = cv.u;
}

// group-id (first-occurrence argmax over channels 3..6) + bucket scatter
__global__ void k_gid(const float* __restrict__ pts, unsigned* __restrict__ cnt,
                      unsigned* __restrict__ perm) {
    int tid = blockIdx.x * blockDim.x + threadIdx.x;
    if (tid >= NB * NPTS) return;
    int b = tid / NPTS, n = tid % NPTS;
    const float* p = pts + (size_t)b * 7 * NPTS + 3 * NPTS + n;
    float v0 = p[0], v1 = p[NPTS], v2 = p[2 * NPTS], v3 = p[3 * NPTS];
    int g = 0; float best = v0;
    if (v1 > best) { best = v1; g = 1; }
    if (v2 > best) { best = v2; g = 2; }
    if (v3 > best) { best = v3; g = 3; }
    unsigned slot = atomicAdd(&cnt[b * 4 + g], 1u);
    perm[(size_t)(b * 4 + g) * NPTS + slot] = (unsigned)n;
}

__global__ void k_g1(const unsigned long long* __restrict__ gacc,
                     const float* __restrict__ wg0, const float* __restrict__ bg0,
                     float* __restrict__ h1) {
    int tid = blockIdx.x * blockDim.x + threadIdx.x;
    if (tid >= 16 * 512) return;
    int o = tid & 511, b = tid >> 9;
    float acc = bg0[o];
    const float* wr = wg0 + (size_t)o * 1024;
    const unsigned long long* ga = gacc + (size_t)b * 1024;
    for (int c = 0; c < 1024; ++c)
        acc += __uint_as_float((unsigned)(ga[c] >> 32)) * wr[c];
    h1[tid] = fmaxf(acc, 0.f);
}

__global__ void k_gr1(const unsigned* __restrict__ grp, const float* __restrict__ wgr0,
                      const float* __restrict__ bgr0, float* __restrict__ hg1) {
    int o = blockIdx.x * blockDim.x + threadIdx.x;
    if (o >= 512) return;
    float acc = bgr0[o];
    const float* wr = wgr0 + (size_t)o * 4096;
    for (int i = 0; i < 4096; ++i) acc += __uint_as_float(grp[i]) * wr[i];
    hg1[o] = fmaxf(acc, 0.f);
}

__global__ void k_g2(const float* __restrict__ h1, const float* __restrict__ wg1,
                     const float* __restrict__ bg1, float* __restrict__ out) {
    int tid = blockIdx.x * blockDim.x + threadIdx.x;
    if (tid >= 16 * 128) return;
    int o = tid & 127, b = tid >> 7;
    float acc = bg1[o];
    const float* wr = wg1 + (size_t)o * 512;
    const float* hr = h1 + (size_t)b * 512;
    for (int k = 0; k < 512; ++k) acc += hr[k] * wr[k];
    out[b * 256 + 128 + o] = fmaxf(acc, 0.f);
}

__global__ void k_gr2(const float* __restrict__ hg1, const float* __restrict__ wgr1,
                      const float* __restrict__ bgr1, float* __restrict__ out) {
    int o = blockIdx.x * blockDim.x + threadIdx.x;
    if (o >= 128) return;
    float acc = bgr1[o];
    const float* wr = wgr1 + (size_t)o * 512;
    for (int k = 0; k < 512; ++k) acc += hg1[k] * wr[k];
    float r = fmaxf(acc, 0.f);
    for (int b = 0; b < 16; ++b) out[b * 256 + o] = r;   // broadcast row
}

__global__ void k_idx(const unsigned long long* __restrict__ gacc, float* __restrict__ out) {
    int i = blockIdx.x * blockDim.x + threadIdx.x;
    if (i >= 16384) return;
    unsigned lo = (unsigned)(gacc[i] & 0xFFFFFFFFull);
    out[4096 + i] = (float)(~lo);                        // index stored as ~n
}

// ---------------- launch ----------------
extern "C" void kernel_launch(void* const* d_in, const int* in_sizes, int n_in,
                              void* d_out, int out_size, void* d_ws, size_t ws_size,
                              hipStream_t stream) {
    const float* pts  = (const float*)d_in[0];
    const float* wl0  = (const float*)d_in[1];  const float* bl0 = (const float*)d_in[2];
    const float* wl1  = (const float*)d_in[3];  const float* bl1 = (const float*)d_in[4];
    const float* wl2  = (const float*)d_in[5];  const float* bl2 = (const float*)d_in[6];
    const float* wl3  = (const float*)d_in[7];  const float* bl3 = (const float*)d_in[8];
    const float* wl4  = (const float*)d_in[9];  const float* bl4 = (const float*)d_in[10];
    const float* wg0  = (const float*)d_in[11]; const float* bg0 = (const float*)d_in[12];
    const float* wg1  = (const float*)d_in[13]; const float* bg1 = (const float*)d_in[14];
    const float* wgr0 = (const float*)d_in[15]; const float* bgr0 = (const float*)d_in[16];
    const float* wgr1 = (const float*)d_in[17]; const float* bgr1 = (const float*)d_in[18];
    float* out = (float*)d_out;

    char* ws = (char*)d_ws;
    unsigned long long* gacc = (unsigned long long*)(ws + 0);        // 16*1024 u64
    unsigned* grp  = (unsigned*)(ws + 131072);                       // 4*1024 u32
    unsigned* cntb = (unsigned*)(ws + 147456);                       // 16*4 u32
    unsigned* perm = (unsigned*)(ws + 147712);                       // 16*4*16384 u32
    unsigned short* wpack = (unsigned short*)(ws + 4342016);         // 149504 halves
    float* h1  = (float*)(ws + 4641024);                             // 16*512 f32
    float* hg1 = (float*)(ws + 4673792);                             // 512 f32

    k_init<<<64, 256, 0, stream>>>(gacc, grp, cntb);
    k_pack<<<8,   256, 0, stream>>>(wl0, wpack + 0,     64,   7, 1);
    k_pack<<<16,  256, 0, stream>>>(wl1, wpack + 2048,  64,  64, 2);
    k_pack<<<16,  256, 0, stream>>>(wl2, wpack + 6144,  64,  64, 2);
    k_pack<<<32,  256, 0, stream>>>(wl3, wpack + 10240, 128, 64, 2);
    k_pack<<<512, 256, 0, stream>>>(wl4, wpack + 18432, 1024, 128, 4);
    k_gid<<<1024, 256, 0, stream>>>(pts, cntb, perm);
    k_main<<<8192, 256, 0, stream>>>(pts, cntb, perm, wpack,
                                     bl0, bl1, bl2, bl3, bl4, gacc, grp);
    k_g1 <<<32, 256, 0, stream>>>(gacc, wg0, bg0, h1);
    k_gr1<<<2,  256, 0, stream>>>(grp, wgr0, bgr0, hg1);
    k_g2 <<<8,  256, 0, stream>>>(h1, wg1, bg1, out);
    k_gr2<<<1,  128, 0, stream>>>(hg1, wgr1, bgr1, out);
    k_idx<<<64, 256, 0, stream>>>(gacc, out);
}
